// SelfAttention_64312840290868
// MI455X (gfx1250) — matrix operand order repaired and checked by reading, not compiled, not run on hardware
//
#include <hip/hip_runtime.h>
#include <hip/hip_bf16.h>
#include <math.h>

typedef __bf16 bf16_t;
typedef __attribute__((ext_vector_type(8)))  __bf16 v8bf;
typedef __attribute__((ext_vector_type(16))) __bf16 v16bf;
typedef __attribute__((ext_vector_type(4)))  __bf16 v4bf;
typedef __attribute__((ext_vector_type(8)))  float  v8f;

#if __has_builtin(__builtin_amdgcn_exp2f)
#define EXP2F __builtin_amdgcn_exp2f
#else
#define EXP2F exp2f
#endif

#define SEQ   2048
#define BSZ   4
#define EMB   768
#define NH    12
#define DH    64
#define NROW  (SEQ * BSZ)          // 8192
#define LOG2E 1.4426950408889634f

// ---------------- WMMA fragment helpers (CDNA5 16x16x32 bf16 layouts) ----------

// A-matrix 16x32 bf16: lane m (0-15): K=0..7 in elems 0..7, K=16..23 in elems 8..15
//                      lane m+16  : K=8..15 and K=24..31.
__device__ __forceinline__ v16bf load_frag_a(const bf16_t* __restrict__ base, int ld) {
  const int lane = threadIdx.x & 31;
  const int m    = lane & 15;
  const int hf   = lane >> 4;
  const bf16_t* p = base + (size_t)m * ld + hf * 8;
  v8bf lo = *(const v8bf*)(p);
  v8bf hi = *(const v8bf*)(p + 16);
  return __builtin_shufflevector(lo, hi, 0,1,2,3,4,5,6,7,8,9,10,11,12,13,14,15);
}

// B-matrix 32x16 bf16 where B[k,n] = S[n,k] (S row-major, ld = row stride):
// lane n (0-15) holds K=0..7 and K=16..23 of column n; lane n+16 holds K=8..15 and K=24..31.
__device__ __forceinline__ v16bf load_frag_b(const bf16_t* __restrict__ base, int ld) {
  const int lane = threadIdx.x & 31;
  const int n    = lane & 15;
  const int hf   = lane >> 4;
  const bf16_t* p = base + (size_t)n * ld + hf * 8;
  v8bf lo = *(const v8bf*)(p);
  v8bf hi = *(const v8bf*)(p + 16);
  return __builtin_shufflevector(lo, hi, 0,1,2,3,4,5,6,7,8,9,10,11,12,13,14,15);
}

__device__ __forceinline__ v8f wmma_bf16(v16bf a, v16bf b, v8f c) {
  return __builtin_amdgcn_wmma_f32_16x16x32_bf16(false, a, false, b, (short)0, c,
                                                 false, false);
}

// Software-pipelined 16x64 GEMM tile over K=EMB: C[16,64] += A[16,768] * S[64,768]^T.
// Two-stage rotated pipeline: while one fragment set feeds the WMMAs, the other
// set is being reloaded in place -> loads stay in flight across the matrix ops
// and there are no loop-carried register copies.
__device__ __forceinline__ void gemm_tile_16x64(const bf16_t* __restrict__ arow,
                                                const bf16_t* __restrict__ sbase,
                                                v8f c[4]) {
  v16bf a0  = load_frag_a(arow, EMB);
  v16bf b00 = load_frag_b(sbase, EMB);
  v16bf b01 = load_frag_b(sbase + (size_t)16 * EMB, EMB);
  v16bf b02 = load_frag_b(sbase + (size_t)32 * EMB, EMB);
  v16bf b03 = load_frag_b(sbase + (size_t)48 * EMB, EMB);
  // 24 K-steps total; 11 loop iterations handle 2 steps each, tail handles 2.
  for (int k0 = 0; k0 < EMB - 64; k0 += 64) {
    v16bf a1  = load_frag_a(arow + k0 + 32, EMB);
    v16bf b10 = load_frag_b(sbase + k0 + 32, EMB);
    v16bf b11 = load_frag_b(sbase + (size_t)16 * EMB + k0 + 32, EMB);
    v16bf b12 = load_frag_b(sbase + (size_t)32 * EMB + k0 + 32, EMB);
    v16bf b13 = load_frag_b(sbase + (size_t)48 * EMB + k0 + 32, EMB);
    c[0] = wmma_bf16(a0, b00, c[0]);
    c[1] = wmma_bf16(a0, b01, c[1]);
    c[2] = wmma_bf16(a0, b02, c[2]);
    c[3] = wmma_bf16(a0, b03, c[3]);
    a0  = load_frag_a(arow + k0 + 64, EMB);
    b00 = load_frag_b(sbase + k0 + 64, EMB);
    b01 = load_frag_b(sbase + (size_t)16 * EMB + k0 + 64, EMB);
    b02 = load_frag_b(sbase + (size_t)32 * EMB + k0 + 64, EMB);
    b03 = load_frag_b(sbase + (size_t)48 * EMB + k0 + 64, EMB);
    c[0] = wmma_bf16(a1, b10, c[0]);
    c[1] = wmma_bf16(a1, b11, c[1]);
    c[2] = wmma_bf16(a1, b12, c[2]);
    c[3] = wmma_bf16(a1, b13, c[3]);
  }
  // tail: stage0 holds k = EMB-64; load and consume k = EMB-32.
  {
    v16bf a1  = load_frag_a(arow + EMB - 32, EMB);
    v16bf b10 = load_frag_b(sbase + EMB - 32, EMB);
    v16bf b11 = load_frag_b(sbase + (size_t)16 * EMB + EMB - 32, EMB);
    v16bf b12 = load_frag_b(sbase + (size_t)32 * EMB + EMB - 32, EMB);
    v16bf b13 = load_frag_b(sbase + (size_t)48 * EMB + EMB - 32, EMB);
    c[0] = wmma_bf16(a0, b00, c[0]);
    c[1] = wmma_bf16(a0, b01, c[1]);
    c[2] = wmma_bf16(a0, b02, c[2]);
    c[3] = wmma_bf16(a0, b03, c[3]);
    c[0] = wmma_bf16(a1, b10, c[0]);
    c[1] = wmma_bf16(a1, b11, c[1]);
    c[2] = wmma_bf16(a1, b12, c[2]);
    c[3] = wmma_bf16(a1, b13, c[3]);
  }
}

// ---------------- f32 -> bf16 conversion ---------------------------------------

__global__ void cvt_f32_bf16_kernel(const float* __restrict__ src,
                                    bf16_t* __restrict__ dst, int n4) {
  int i = blockIdx.x * blockDim.x + threadIdx.x;
  if (i < n4) {
    const float4 f = ((const float4*)src)[i];
    v4bf o;
    o[0] = (bf16_t)f.x; o[1] = (bf16_t)f.y; o[2] = (bf16_t)f.z; o[3] = (bf16_t)f.w;
    ((v4bf*)dst)[i] = o;
  }
}

// ---------------- QKV projection: [8192,768] x W^T, heads laid out for attention

__global__ void __launch_bounds__(128, 1)
qkv_proj_kernel(const bf16_t* __restrict__ xb,
                const bf16_t* __restrict__ wq,
                const bf16_t* __restrict__ wk,
                const bf16_t* __restrict__ wv,
                const float* __restrict__ bq,
                const float* __restrict__ bk,
                const float* __restrict__ bv,
                bf16_t* __restrict__ qo,   // [bh, t, d]
                bf16_t* __restrict__ ko,   // [bh, t, d]
                bf16_t* __restrict__ vto)  // [bh, d, t]
{
  const int wid = blockIdx.x * 4 + (threadIdx.x >> 5);
  const int mt  = wid & 511;               // 512 row tiles of 16
  const int ns  = (wid >> 9) % 12;         // 12 col super-tiles of 64
  const int mat = wid / (512 * 12);        // 0=q 1=k 2=v

  const bf16_t* W    = (mat == 0) ? wq : (mat == 1) ? wk : wv;
  const float*  bias = (mat == 0) ? bq : (mat == 1) ? bk : bv;

  v8f acc[4] = {};
  gemm_tile_16x64(xb + (size_t)mt * 16 * EMB, W + (size_t)ns * 64 * EMB, acc);

  const int lane = threadIdx.x & 31;
  const int ln   = lane & 15;
  const int hf   = lane >> 4;
#pragma unroll
  for (int ct = 0; ct < 4; ++ct) {
    const int e  = ns * 64 + ct * 16 + ln;
    const float bv_ = bias[e];
    const int h = e >> 6, d = e & 63;
#pragma unroll
    for (int r = 0; r < 8; ++r) {
      const int m  = mt * 16 + r + hf * 8;     // row = t*BSZ + b
      const int t  = m >> 2;
      const int b  = m & 3;
      const int bh = b * NH + h;
      float val = acc[ct][r] + bv_;
      if (mat == 0)      qo[((size_t)bh * SEQ + t) * DH + d] = (bf16_t)(val * 0.125f);
      else if (mat == 1) ko[((size_t)bh * SEQ + t) * DH + d] = (bf16_t)val;
      else               vto[((size_t)bh * DH + d) * SEQ + t] = (bf16_t)val;
    }
  }
}

// ---------------- streaming-softmax attention: one wave per (bh, 16 query rows)

__global__ void __launch_bounds__(128, 1)
attn_kernel(const bf16_t* __restrict__ qb,
            const bf16_t* __restrict__ kb,
            const bf16_t* __restrict__ vtb,
            bf16_t* __restrict__ attnb)  // [t*BSZ+b, 768] bf16
{
  __shared__ __align__(16) bf16_t plds[4][16 * 32];   // 1 KB per wave
  const int w   = threadIdx.x >> 5;
  const int wid = blockIdx.x * 4 + w;
  const int bh  = wid >> 7;            // 48 head-batches
  const int it  = wid & 127;           // 128 query tiles

  const bf16_t* Qb = qb + ((size_t)bh * SEQ + it * 16) * DH;
  const bf16_t* Kb = kb + (size_t)bh * SEQ * DH;
  const bf16_t* Vt = vtb + (size_t)bh * DH * SEQ;
  bf16_t* pl = &plds[w][0];

  const v16bf qf0 = load_frag_a(Qb, DH);
  const v16bf qf1 = load_frag_a(Qb + 32, DH);

  v8f o0 = {}, o1 = {}, o2 = {}, o3 = {};
  float mrow[8], srow[8];
#pragma unroll
  for (int r = 0; r < 8; ++r) { mrow[r] = -3.0e38f; srow[r] = 0.0f; }

  const int lane = threadIdx.x & 31;
  const int ln   = lane & 15;
  const int hf   = lane >> 4;

  for (int j0 = 0; j0 < SEQ; j0 += 32) {
    // Issue ALL vmem for this step up front: 4 K-fragments for the score WMMAs
    // and 4 V-fragments that are only consumed after softmax + LDS relayout,
    // so the V loads overlap the score WMMAs and the softmax VALU work.
    const v16bf kb0 = load_frag_b(Kb + (size_t)j0 * DH, DH);
    const v16bf kb1 = load_frag_b(Kb + (size_t)j0 * DH + 32, DH);
    const v16bf kb2 = load_frag_b(Kb + (size_t)(j0 + 16) * DH, DH);
    const v16bf kb3 = load_frag_b(Kb + (size_t)(j0 + 16) * DH + 32, DH);
    const v16bf vb0 = load_frag_b(Vt + (size_t) 0 * SEQ + j0, SEQ);
    const v16bf vb1 = load_frag_b(Vt + (size_t)16 * SEQ + j0, SEQ);
    const v16bf vb2 = load_frag_b(Vt + (size_t)32 * SEQ + j0, SEQ);
    const v16bf vb3 = load_frag_b(Vt + (size_t)48 * SEQ + j0, SEQ);

    // Prefetch next step's K tile (32 rows x 128B) and V tile (64 x 64B
    // strided segments) into cache: global_prefetch_b8, one line per lane.
    const int jn = (j0 + 32 < SEQ) ? (j0 + 32) : j0;
    __builtin_prefetch(Kb + (size_t)(jn + lane) * DH, 0, 3);
    __builtin_prefetch(Vt + (size_t)lane * SEQ + jn, 0, 3);
    __builtin_prefetch(Vt + (size_t)(32 + lane) * SEQ + jn, 0, 3);

    // scores for two 16-wide key tiles, f32 accumulate over Dh=64
    v8f s0 = {}, s1 = {};
    s0 = wmma_bf16(qf0, kb0, s0);
    s0 = wmma_bf16(qf1, kb1, s0);
    s1 = wmma_bf16(qf0, kb2, s1);
    s1 = wmma_bf16(qf1, kb3, s1);

    // online softmax in C-fragment layout (row r+8*hf spread over 16 lanes)
#pragma unroll
    for (int r = 0; r < 8; ++r) {
      float tm = fmaxf(s0[r], s1[r]);
      tm = fmaxf(tm, __shfl_xor(tm, 1, 16));
      tm = fmaxf(tm, __shfl_xor(tm, 2, 16));
      tm = fmaxf(tm, __shfl_xor(tm, 4, 16));
      tm = fmaxf(tm, __shfl_xor(tm, 8, 16));
      const float mn   = fmaxf(mrow[r], tm);
      const float corr = EXP2F((mrow[r] - mn) * LOG2E);
      const float p0   = EXP2F((s0[r] - mn) * LOG2E);
      const float p1   = EXP2F((s1[r] - mn) * LOG2E);
      mrow[r] = mn;
      float ps = p0 + p1;
      ps += __shfl_xor(ps, 1, 16);
      ps += __shfl_xor(ps, 2, 16);
      ps += __shfl_xor(ps, 4, 16);
      ps += __shfl_xor(ps, 8, 16);
      srow[r] = srow[r] * corr + ps;
      o0[r] *= corr; o1[r] *= corr; o2[r] *= corr; o3[r] *= corr;
      const int row = r + hf * 8;
      pl[row * 32 + ln]      = (bf16_t)p0;
      pl[row * 32 + 16 + ln] = (bf16_t)p1;
    }

    // relayout P (C-frag -> A-frag) via per-wave LDS slab, then P·V
    const v16bf pa = load_frag_a(pl, 32);
    o0 = wmma_bf16(pa, vb0, o0);
    o1 = wmma_bf16(pa, vb1, o1);
    o2 = wmma_bf16(pa, vb2, o2);
    o3 = wmma_bf16(pa, vb3, o3);
  }

  const int b = bh / NH, h = bh % NH;
#pragma unroll
  for (int r = 0; r < 8; ++r) {
    const int t = it * 16 + r + hf * 8;
    const float inv = 1.0f / srow[r];
    const size_t rowoff = (size_t)(t * BSZ + b) * EMB + h * DH + ln;
    attnb[rowoff +  0] = (bf16_t)(o0[r] * inv);
    attnb[rowoff + 16] = (bf16_t)(o1[r] * inv);
    attnb[rowoff + 32] = (bf16_t)(o2[r] * inv);
    attnb[rowoff + 48] = (bf16_t)(o3[r] * inv);
  }
}

// ---------------- output projection: attn[8192,768] x Wout^T + bout -> f32 -----

__global__ void __launch_bounds__(128, 1)
out_proj_kernel(const bf16_t* __restrict__ attnb,
                const bf16_t* __restrict__ wo,
                const float* __restrict__ bo,
                float* __restrict__ out)
{
  const int wid = blockIdx.x * 4 + (threadIdx.x >> 5);
  const int mt  = wid & 511;
  const int ns  = wid >> 9;

  v8f acc[4] = {};
  gemm_tile_16x64(attnb + (size_t)mt * 16 * EMB, wo + (size_t)ns * 64 * EMB, acc);

  const int lane = threadIdx.x & 31;
  const int ln   = lane & 15;
  const int hf   = lane >> 4;
#pragma unroll
  for (int ct = 0; ct < 4; ++ct) {
    const int e = ns * 64 + ct * 16 + ln;
    const float bv_ = bo[e];
#pragma unroll
    for (int r = 0; r < 8; ++r) {
      const int m = mt * 16 + r + hf * 8;
      out[(size_t)m * EMB + e] = acc[ct][r] + bv_;   // coalesced b32 stores
    }
  }
}

// ---------------- host-side launch ---------------------------------------------

extern "C" void kernel_launch(void* const* d_in, const int* in_sizes, int n_in,
                              void* d_out, int out_size, void* d_ws, size_t ws_size,
                              hipStream_t stream) {
  const float* x    = (const float*)d_in[0];
  const float* Wq   = (const float*)d_in[1];
  const float* bq   = (const float*)d_in[2];
  const float* Wk   = (const float*)d_in[3];
  const float* bk   = (const float*)d_in[4];
  const float* Wv   = (const float*)d_in[5];
  const float* bv   = (const float*)d_in[6];
  const float* Wout = (const float*)d_in[7];
  const float* bout = (const float*)d_in[8];
  float* out = (float*)d_out;

  char* ws = (char*)d_ws;
  const size_t XB = (size_t)NROW * EMB * 2;          // 12,582,912
  const size_t WB = (size_t)EMB * EMB * 2;           //  1,179,648
  bf16_t* xb   = (bf16_t*)(ws);
  bf16_t* wqb  = (bf16_t*)(ws + XB);
  bf16_t* wkb  = (bf16_t*)(ws + XB + WB);
  bf16_t* wvb  = (bf16_t*)(ws + XB + 2 * WB);
  bf16_t* wob  = (bf16_t*)(ws + XB + 3 * WB);
  bf16_t* qb   = (bf16_t*)(ws + XB + 4 * WB);
  bf16_t* kb   = (bf16_t*)(ws + 2 * XB + 4 * WB);
  bf16_t* vtb  = (bf16_t*)(ws + 3 * XB + 4 * WB);
  bf16_t* atb  = (bf16_t*)(ws + 4 * XB + 4 * WB);
  // total ws use: 5*XB + 4*WB = 67,633,152 bytes

  // f32 -> bf16 conversions
  cvt_f32_bf16_kernel<<<(NROW * EMB / 4 + 255) / 256, 256, 0, stream>>>(x, xb, NROW * EMB / 4);
  cvt_f32_bf16_kernel<<<(EMB * EMB / 4 + 255) / 256, 256, 0, stream>>>(Wq, wqb, EMB * EMB / 4);
  cvt_f32_bf16_kernel<<<(EMB * EMB / 4 + 255) / 256, 256, 0, stream>>>(Wk, wkb, EMB * EMB / 4);
  cvt_f32_bf16_kernel<<<(EMB * EMB / 4 + 255) / 256, 256, 0, stream>>>(Wv, wvb, EMB * EMB / 4);
  cvt_f32_bf16_kernel<<<(EMB * EMB / 4 + 255) / 256, 256, 0, stream>>>(Wout, wob, EMB * EMB / 4);

  // QKV projections: 512 m-tiles * 12 n-supertiles * 3 matrices, 4 waves/block
  qkv_proj_kernel<<<512 * 12 * 3 / 4, 128, 0, stream>>>(xb, wqb, wkb, wvb,
                                                        bq, bk, bv, qb, kb, vtb);

  // attention: 48 head-batches * 128 query tiles, 4 waves/block
  attn_kernel<<<48 * 128 / 4, 128, 0, stream>>>(qb, kb, vtb, atb);

  // output projection
  out_proj_kernel<<<512 * 12 / 4, 128, 0, stream>>>(atb, wob, bout, out);
}
